// self_attention_69733089017887
// MI455X (gfx1250) — matrix operand order repaired and checked by reading, not compiled, not run on hardware
//
#include <hip/hip_runtime.h>
#include <math.h>

// Problem constants (hard-coded in reference)
#define Hh      8
#define Ll      384
#define Dd      1024
#define BATCH   8
#define DHd     1024
#define HDd     8192      // H*DH
#define BLn     3072      // BATCH*L
#define PREFIXn 257

typedef __attribute__((ext_vector_type(16))) __bf16 bf16x16;
typedef __attribute__((ext_vector_type(8)))  float  v8f;
typedef unsigned int  u32x4 __attribute__((ext_vector_type(4)));
typedef unsigned short u16;

__device__ __forceinline__ u16 f2bf(float f) {
  unsigned u = __float_as_uint(f);
  u += 0x7FFFu + ((u >> 16) & 1u);          // round-to-nearest-even
  return (u16)(u >> 16);
}

// A fragment: 16x32 bf16, lane(&15)=M row, lane>>4 selects K-interleave of 8.
// Source is row-major [M, K] with leading dim ld (elements).
__device__ __forceinline__ bf16x16 ld_a_frag(const u16* p, int lane, int ld) {
  const u16* q = p + (size_t)(lane & 15) * ld + ((lane >> 4) << 3);
  union { bf16x16 v; u32x4 u[2]; } r;
  r.u[0] = *(const u32x4*)q;
  r.u[1] = *(const u32x4*)(q + 16);
  return r.v;
}

// B fragment: 32x16 bf16, lane(&15)=N row of row-major [N, K] source,
// lane>>4 selects K-interleave of 8 (same as A).
__device__ __forceinline__ bf16x16 ld_b_frag(const u16* p, int lane, int ld) {
  const u16* q = p + (size_t)(lane & 15) * ld + ((lane >> 4) << 3);
  union { bf16x16 v; u32x4 u[2]; } r;
  r.u[0] = *(const u32x4*)q;
  r.u[1] = *(const u32x4*)(q + 16);
  return r.v;
}

// Wave computes a 64x64 f32 tile of A[M,K] x B[N,K]^T (both bf16 row-major).
__device__ __forceinline__ void wmma_core_64x64(const u16* __restrict__ Aw, int lda,
                                                const u16* __restrict__ Bw, int ldb,
                                                int K, int lane, v8f acc[4][4]) {
  for (int k = 0; k < K; k += 32) {
    if (k + 32 < K) {
      __builtin_prefetch(Aw + (size_t)(lane & 15) * lda + k + 32, 0, 1);
      __builtin_prefetch(Bw + (size_t)(lane & 15) * ldb + k + 32, 0, 1);
    }
    bf16x16 af[4], bfr[4];
#pragma unroll
    for (int i = 0; i < 4; ++i)
      af[i] = ld_a_frag(Aw + (size_t)(i * 16) * lda + k, lane, lda);
#pragma unroll
    for (int i = 0; i < 4; ++i)
      bfr[i] = ld_b_frag(Bw + (size_t)(i * 16) * ldb + k, lane, ldb);
#pragma unroll
    for (int mi = 0; mi < 4; ++mi)
#pragma unroll
      for (int ni = 0; ni < 4; ++ni)
        acc[mi][ni] = __builtin_amdgcn_wmma_f32_16x16x32_bf16(
            false, af[mi], false, bfr[ni], (short)0, acc[mi][ni], false, false);
  }
}

#define GEMM_PROLOGUE                                                     \
  int tid = threadIdx.x, lane = tid & 31, wave = tid >> 5;                \
  int m0 = blockIdx.x * 128 + (wave >> 1) * 64;                           \
  int n0 = blockIdx.y * 128 + (wave & 1) * 64;                            \
  const v8f vzero = {0.f,0.f,0.f,0.f,0.f,0.f,0.f,0.f};                    \
  v8f acc[4][4];                                                          \
  _Pragma("unroll") for (int i_ = 0; i_ < 4; ++i_)                        \
  _Pragma("unroll") for (int j_ = 0; j_ < 4; ++j_) acc[i_][j_] = vzero;

// -------- fp32 -> bf16 convert --------
__global__ __launch_bounds__(256) void k_cvt(const float* __restrict__ in,
                                             u16* __restrict__ outp, size_t n) {
  size_t i = (size_t)blockIdx.x * blockDim.x + threadIdx.x;
  size_t stride = (size_t)gridDim.x * blockDim.x;
  for (; i < n; i += stride) outp[i] = f2bf(in[i]);
}

// -------- generic NT GEMM, bf16 out --------
__global__ __launch_bounds__(128) void k_gemm_nt_bf16(
    const u16* __restrict__ A, int lda, const u16* __restrict__ B, int ldb,
    u16* __restrict__ C, int ldc, int K) {
  GEMM_PROLOGUE
  wmma_core_64x64(A + (size_t)m0 * lda, lda, B + (size_t)n0 * ldb, ldb, K, lane, acc);
  int nl = lane & 15, mh = (lane >> 4) << 3;
#pragma unroll
  for (int mi = 0; mi < 4; ++mi)
#pragma unroll
    for (int ni = 0; ni < 4; ++ni) {
      int n = n0 + ni * 16 + nl;
      int mbase = m0 + mi * 16 + mh;
#pragma unroll
      for (int r = 0; r < 8; ++r)
        C[(size_t)(mbase + r) * ldc + n] = f2bf(acc[mi][ni][r]);
    }
}

// -------- generic NT GEMM, f32 out --------
__global__ __launch_bounds__(128) void k_gemm_nt_f32(
    const u16* __restrict__ A, int lda, const u16* __restrict__ B, int ldb,
    float* __restrict__ C, int ldc, int K) {
  GEMM_PROLOGUE
  wmma_core_64x64(A + (size_t)m0 * lda, lda, B + (size_t)n0 * ldb, ldb, K, lane, acc);
  int nl = lane & 15, mh = (lane >> 4) << 3;
#pragma unroll
  for (int mi = 0; mi < 4; ++mi)
#pragma unroll
    for (int ni = 0; ni < 4; ++ni) {
      int n = n0 + ni * 16 + nl;
      int mbase = m0 + mi * 16 + mh;
#pragma unroll
      for (int r = 0; r < 8; ++r)
        C[(size_t)(mbase + r) * ldc + n] = acc[mi][ni][r];
    }
}

// -------- V projection with per-head transposed store: VT[b][h][d][l] --------
__global__ __launch_bounds__(128) void k_gemm_vt(const u16* __restrict__ Xb,
                                                 const u16* __restrict__ Wvb,
                                                 u16* __restrict__ VT) {
  GEMM_PROLOGUE
  wmma_core_64x64(Xb + (size_t)m0 * Dd, Dd, Wvb + (size_t)n0 * Dd, Dd, Dd, lane, acc);
  int nl = lane & 15, mh = (lane >> 4) << 3;
#pragma unroll
  for (int mi = 0; mi < 4; ++mi)
#pragma unroll
    for (int ni = 0; ni < 4; ++ni) {
      int n = n0 + ni * 16 + nl;        // h*DH + d
      int m = m0 + mi * 16 + mh;        // b*L + l  (8 consecutive l)
      int b = m / Ll, l = m - b * Ll;
      u16* dst = VT + ((size_t)(b * Hh + (n >> 10)) * DHd + (n & (DHd - 1))) * Ll + l;
      u32x4 pk;
      pk.x = (unsigned)f2bf(acc[mi][ni][0]) | ((unsigned)f2bf(acc[mi][ni][1]) << 16);
      pk.y = (unsigned)f2bf(acc[mi][ni][2]) | ((unsigned)f2bf(acc[mi][ni][3]) << 16);
      pk.z = (unsigned)f2bf(acc[mi][ni][4]) | ((unsigned)f2bf(acc[mi][ni][5]) << 16);
      pk.w = (unsigned)f2bf(acc[mi][ni][6]) | ((unsigned)f2bf(acc[mi][ni][7]) << 16);
      *(u32x4*)dst = pk;
    }
}

// -------- S = Q K^T * scale with pairwise mask, f32 out, one (b,h) per blockIdx.z ----
__global__ __launch_bounds__(128) void k_gemm_score(const u16* __restrict__ Qb,
                                                    const u16* __restrict__ Kb,
                                                    const float* __restrict__ am,
                                                    float* __restrict__ S) {
  int bh = blockIdx.z, b = bh >> 3, h = bh & 7;
  const u16* Abase = Qb + (size_t)(b * Ll) * HDd + h * DHd;
  const u16* Bbase = Kb + (size_t)(b * Ll) * HDd + h * DHd;
  float* Sp = S + (size_t)bh * Ll * Ll;
  GEMM_PROLOGUE
  wmma_core_64x64(Abase + (size_t)m0 * HDd, HDd, Bbase + (size_t)n0 * HDd, HDd, DHd, lane, acc);
  int nl = lane & 15, mh = (lane >> 4) << 3;
#pragma unroll
  for (int mi = 0; mi < 4; ++mi)
#pragma unroll
    for (int ni = 0; ni < 4; ++ni) {
      int j = n0 + ni * 16 + nl;
      float mj = (j < PREFIXn) ? 1.0f : am[b * (Ll - PREFIXn) + (j - PREFIXn)];
      int mbase = m0 + mi * 16 + mh;
#pragma unroll
      for (int r = 0; r < 8; ++r) {
        int i = mbase + r;
        float mi_ = (i < PREFIXn) ? 1.0f : am[b * (Ll - PREFIXn) + (i - PREFIXn)];
        float v = ((mi_ + mj) == 2.0f) ? acc[mi][ni][r] * 0.03125f : -INFINITY;
        Sp[(size_t)i * Ll + j] = v;
      }
    }
}

// -------- row softmax over L=384, f32 in -> bf16 out --------
__global__ __launch_bounds__(128) void k_softmax(const float* __restrict__ S,
                                                 u16* __restrict__ P) {
  int row = blockIdx.x, tid = threadIdx.x;
  const float* s = S + (size_t)row * Ll;
  u16* p = P + (size_t)row * Ll;
  float v0 = s[tid], v1 = s[tid + 128], v2 = s[tid + 256];
  __shared__ float red[128];
  float mx = fmaxf(v0, fmaxf(v1, v2));
  red[tid] = mx; __syncthreads();
  for (int st = 64; st > 0; st >>= 1) {
    if (tid < st) red[tid] = fmaxf(red[tid], red[tid + st]);
    __syncthreads();
  }
  mx = red[0]; __syncthreads();
  float e0 = __expf(v0 - mx), e1 = __expf(v1 - mx), e2 = __expf(v2 - mx);
  red[tid] = e0 + e1 + e2; __syncthreads();
  for (int st = 64; st > 0; st >>= 1) {
    if (tid < st) red[tid] += red[tid + st];
    __syncthreads();
  }
  float inv = 1.0f / red[0];
  p[tid] = f2bf(e0 * inv); p[tid + 128] = f2bf(e1 * inv); p[tid + 256] = f2bf(e2 * inv);
}

// -------- O = P @ V per (b,h), NT against pre-transposed VT, bf16 out [B*L, HD] ----
__global__ __launch_bounds__(128) void k_gemm_pv(const u16* __restrict__ P,
                                                 const u16* __restrict__ VT,
                                                 u16* __restrict__ O) {
  int bh = blockIdx.z, b = bh >> 3, h = bh & 7;
  const u16* Abase = P + (size_t)bh * Ll * Ll;     // [L, L]
  const u16* Bbase = VT + (size_t)bh * DHd * Ll;   // [DH, L]
  GEMM_PROLOGUE
  wmma_core_64x64(Abase + (size_t)m0 * Ll, Ll, Bbase + (size_t)n0 * Ll, Ll, Ll, lane, acc);
  int nl = lane & 15, mh = (lane >> 4) << 3;
#pragma unroll
  for (int mi = 0; mi < 4; ++mi)
#pragma unroll
    for (int ni = 0; ni < 4; ++ni) {
      int d = n0 + ni * 16 + nl;
      int qb = m0 + mi * 16 + mh;
#pragma unroll
      for (int r = 0; r < 8; ++r)
        O[(size_t)(b * Ll + qb + r) * HDd + h * DHd + d] = f2bf(acc[mi][ni][r]);
    }
}

// -------- residual add + LayerNorm, f32 out --------
__global__ __launch_bounds__(256) void k_ln(const float* __restrict__ U,
                                            const float* __restrict__ X,
                                            const float* __restrict__ w,
                                            const float* __restrict__ bb,
                                            float* __restrict__ out) {
  int row = blockIdx.x, tid = threadIdx.x;
  const float* u = U + (size_t)row * Dd;
  const float* x = X + (size_t)row * Dd;
  float v[4], s = 0.f, ss = 0.f;
#pragma unroll
  for (int i = 0; i < 4; ++i) {
    int c = tid + i * 256;
    float t = u[c] + x[c];
    v[i] = t; s += t; ss += t * t;
  }
  __shared__ float rs[256], rq[256];
  rs[tid] = s; rq[tid] = ss; __syncthreads();
  for (int st = 128; st > 0; st >>= 1) {
    if (tid < st) { rs[tid] += rs[tid + st]; rq[tid] += rq[tid + st]; }
    __syncthreads();
  }
  float mean = rs[0] * (1.0f / 1024.0f);
  float var  = rq[0] * (1.0f / 1024.0f) - mean * mean;
  float inv  = rsqrtf(var + 1e-5f);
  float* o = out + (size_t)row * Dd;
#pragma unroll
  for (int i = 0; i < 4; ++i) {
    int c = tid + i * 256;
    o[c] = (v[i] - mean) * inv * w[c] + bb[c];
  }
}

extern "C" void kernel_launch(void* const* d_in, const int* in_sizes, int n_in,
                              void* d_out, int out_size, void* d_ws, size_t ws_size,
                              hipStream_t stream) {
  const float* X   = (const float*)d_in[0];
  const float* am  = (const float*)d_in[1];
  const float* Wq  = (const float*)d_in[2];
  const float* Wk  = (const float*)d_in[3];
  const float* Wv  = (const float*)d_in[4];
  const float* Wo  = (const float*)d_in[5];
  const float* lnw = (const float*)d_in[6];
  const float* lnb = (const float*)d_in[7];
  float* out = (float*)d_out;

  char* ws = (char*)d_ws;
  size_t off = 0;
  auto alloc = [&](size_t bytes) -> char* {
    char* p = ws + off;
    off += (bytes + 255) & ~(size_t)255;
    return p;
  };
  u16*  Xb  = (u16*)alloc((size_t)BLn * Dd * 2);       //  6.3 MB
  u16*  Wqb = (u16*)alloc((size_t)HDd * Dd * 2);       // 16.8 MB
  u16*  Wkb = (u16*)alloc((size_t)HDd * Dd * 2);
  u16*  Wvb = (u16*)alloc((size_t)HDd * Dd * 2);
  u16*  Wob = (u16*)alloc((size_t)Dd * HDd * 2);
  u16*  Qb  = (u16*)alloc((size_t)BLn * HDd * 2);      // 50.3 MB
  u16*  Kb  = (u16*)alloc((size_t)BLn * HDd * 2);
  u16*  VTb = (u16*)alloc((size_t)BATCH * Hh * DHd * Ll * 2);
  float* Sb = (float*)alloc((size_t)BATCH * Hh * Ll * Ll * 4);  // 37.7 MB (fits L2)
  u16*  Pb  = (u16*)alloc((size_t)BATCH * Hh * Ll * Ll * 2);
  u16*  Ob  = (u16*)alloc((size_t)BLn * HDd * 2);
  float* Ub = (float*)alloc((size_t)BLn * Dd * 4);
  if (ws_size < off) return;   // workspace too small: bail deterministically

  // 1) fp32 -> bf16 conversions (bandwidth-trivial)
  k_cvt<<<dim3(1024), 256, 0, stream>>>(X,  Xb,  (size_t)BLn * Dd);
  k_cvt<<<dim3(2048), 256, 0, stream>>>(Wq, Wqb, (size_t)HDd * Dd);
  k_cvt<<<dim3(2048), 256, 0, stream>>>(Wk, Wkb, (size_t)HDd * Dd);
  k_cvt<<<dim3(2048), 256, 0, stream>>>(Wv, Wvb, (size_t)HDd * Dd);
  k_cvt<<<dim3(2048), 256, 0, stream>>>(Wo, Wob, (size_t)Dd * HDd);

  // 2) QKV projections: [3072,1024] x [8192,1024]^T
  k_gemm_nt_bf16<<<dim3(24, 64), 128, 0, stream>>>(Xb, Dd, Wqb, Dd, Qb, HDd, Dd);
  k_gemm_nt_bf16<<<dim3(24, 64), 128, 0, stream>>>(Xb, Dd, Wkb, Dd, Kb, HDd, Dd);
  k_gemm_vt    <<<dim3(24, 64), 128, 0, stream>>>(Xb, Wvb, VTb);   // stores [b,h,d,l]

  // 3) scores = Q K^T / 32 + mask  (per head; 37.7 MB total, lives in L2)
  k_gemm_score<<<dim3(3, 3, 64), 128, 0, stream>>>(Qb, Kb, am, Sb);

  // 4) row softmax -> bf16 probabilities
  k_softmax<<<dim3(BATCH * Hh * Ll), 128, 0, stream>>>(Sb, Pb);

  // 5) O = P V  (NT against transposed V)
  k_gemm_pv<<<dim3(3, 8, 64), 128, 0, stream>>>(Pb, VTb, Ob);

  // 6) output projection: [3072,8192] x [1024,8192]^T -> f32
  k_gemm_nt_f32<<<dim3(24, 8), 128, 0, stream>>>(Ob, HDd, Wob, HDd, Ub, Dd, HDd);

  // 7) residual + LayerNorm
  k_ln<<<dim3(BLn), 256, 0, stream>>>(Ub, X, lnw, lnb, out);
}
